// Text2SemanticDecoder_11759620456880
// MI455X (gfx1250) — matrix operand order repaired and checked by reading, not compiled, not run on hardware
//
#include <hip/hip_runtime.h>
#include <hip/hip_bf16.h>

// ---------------------------------------------------------------------------
// Transformer decoder forward (24 layers, D=512, H=16, hd=32, FF=2048, S=2048)
// MI455X / gfx1250: wave32, WMMA f32<-f16 matrix cores, TDM tile staging.
// ---------------------------------------------------------------------------

typedef __attribute__((ext_vector_type(16))) _Float16 v16h;
typedef __attribute__((ext_vector_type(8)))  float    v8f;
typedef unsigned int u32x4 __attribute__((ext_vector_type(4)));
typedef int          i32x4 __attribute__((ext_vector_type(4)));
typedef int          i32x8 __attribute__((ext_vector_type(8)));

#define S_TOT 2048
#define D_MODEL 512
#define N_HEAD 16
#define HEAD_DIM 32
#define FF_DIM 2048
#define N_LAYER 24
#define X_LEN 1024          // reference x_len (compile-time constant of the ref)
#define LN_EPS 1e-5f

#ifndef __has_builtin
#define __has_builtin(x) 0
#endif
#if __has_builtin(__builtin_amdgcn_tensor_load_to_lds) && \
    __has_builtin(__builtin_amdgcn_s_wait_tensorcnt)
#define HAVE_TDM 1
#else
#define HAVE_TDM 0
#endif

#if HAVE_TDM
// ---------------------------------------------------------------------------
// Issue one TDM 2D tile load: tile_h lines x tile_w f32 elements from a
// row-major tensor (line stride row_stride elems) into contiguous LDS.
// Per CDNA5 ISA D# layout:
//   group0: count=1 | lds_addr | global_addr[56:0] | type=2
//   group1: data_size=2 (4B) | tensor_dim0/1 | tile_dim0/1 | dim0_stride
//   groups 2/3: zero (2D tile, no iterate/gather)
// Wave-level op: EXEC ignored, tracked by TENSORcnt.
// ---------------------------------------------------------------------------
__device__ __forceinline__ void tdm_load_2d_f32(
    unsigned lds_byte_off, const float* gptr,
    unsigned tensor_w, unsigned tensor_h,
    unsigned long long row_stride_elems,
    unsigned tile_w, unsigned tile_h) {
  unsigned long long ga = (unsigned long long)(uintptr_t)gptr;
  u32x4 g0;
  g0[0] = 1u;                                       // count=1 (valid, user mode)
  g0[1] = lds_byte_off;                             // lds_addr
  g0[2] = (unsigned)(ga & 0xffffffffu);             // global_addr[31:0]
  g0[3] = (unsigned)((ga >> 32) & 0x01ffffffu)      // global_addr[56:32]
          | (2u << 30);                             // type = 2 ("image")
  i32x8 g1;
  g1[0] = (int)(2u << 16);                          // data_size=2 (4 bytes)
  g1[1] = (int)((tensor_w & 0xffffu) << 16);        // tensor_dim0[15:0]
  g1[2] = (int)((tensor_w >> 16) |
                ((tensor_h & 0xffffu) << 16));      // dim0[31:16] | dim1[15:0]
  g1[3] = (int)((tensor_h >> 16) |
                ((tile_w & 0xffffu) << 16));        // dim1[31:16] | tile_dim0
  g1[4] = (int)(tile_h & 0xffffu);                  // tile_dim1 (tile_dim2=0)
  g1[5] = (int)(unsigned)(row_stride_elems & 0xffffffffu);  // dim0_stride lo
  g1[6] = (int)(unsigned)((row_stride_elems >> 32) & 0xffffu); // dim0_stride hi
  g1[7] = 0;                                        // dim1_stride (unused, 2D)
  i32x4 g2 = {0, 0, 0, 0};
  i32x4 g3 = {0, 0, 0, 0};
#if __clang_major__ >= 23
  i32x8 g4 = {0, 0, 0, 0, 0, 0, 0, 0};
  __builtin_amdgcn_tensor_load_to_lds(g0, g1, g2, g3, g4, 0);
#else
  __builtin_amdgcn_tensor_load_to_lds(g0, g1, g2, g3, 0);
#endif
}
#endif  // HAVE_TDM

// ===========================================================================
// Generic GEMM:  C[M,N] = act( A[M,K] @ W[N,K]^T + bias[N] )
// Block tile 128(M) x 128(N), K-step 32, 256 threads = 8 waves (4x2).
// Each wave owns a 32x64 tile = 2x4 v_wmma_f32_16x16x32_f16 per K-step.
// Operand tiles staged f32 in LDS by the Tensor Data Mover (wave 0 issues,
// waits on TENSORcnt, workgroup barrier publishes); f32->f16 conversion
// happens during fragment build. All dims divide tiles exactly -> EXEC all-1s.
// ===========================================================================
#define BM 128
#define BN 128
#define BK 32

template <bool RELU>
__global__ __launch_bounds__(256) void gemm_wmma(
    const float* __restrict__ A,     // [M,K]
    const float* __restrict__ W,     // [N,K] (compute A @ W^T)
    const float* __restrict__ bias,  // [N]
    float* __restrict__ C,           // [M,N]
    int M, int N, int K) {
  __shared__ float sA[BM * BK];   // 16 KB
  __shared__ float sW[BN * BK];   // 16 KB

  const int tid  = threadIdx.x;
  const int lane = tid & 31;
  const int wave = tid >> 5;   // 0..7
  const int wm   = wave >> 1;  // 0..3  (M sub-tile, 32 rows)
  const int wn   = wave & 1;   // 0..1  (N sub-tile, 64 cols)

  const int blockM = blockIdx.y * BM;
  const int blockN = blockIdx.x * BN;

  v8f acc[2][4] = {};  // [mi][ni] 16x16 f32 accumulators

  const int kofsA = (lane < 16) ? 0 : 8;   // A-operand K-half per lane group
  const int kofsB = (lane < 16) ? 0 : 16;  // unused; B fragments also use kofsA
  const int mrow  = lane & 15;
  const int rbase = (lane < 16) ? 0 : 8;

  const int kTiles = K / BK;
  for (int kt = 0; kt < kTiles; ++kt) {
    const int k0 = kt * BK;

#if HAVE_TDM
    if (tid < 32) {  // wave 0 drives the Tensor Data Mover
      tdm_load_2d_f32((unsigned)(uintptr_t)&sA[0],
                      A + (size_t)blockM * K + k0,
                      (unsigned)K, (unsigned)M, (unsigned long long)K,
                      BK, BM);
      tdm_load_2d_f32((unsigned)(uintptr_t)&sW[0],
                      W + (size_t)blockN * K + k0,
                      (unsigned)K, (unsigned)N, (unsigned long long)K,
                      BK, BN);
      __builtin_amdgcn_s_wait_tensorcnt(0);
    }
#else
    // fallback: cooperative VGPR staging (16 elems/thread per tile)
#pragma unroll
    for (int i = 0; i < 16; ++i) {
      int idx = tid + i * 256;  // 0..4095
      int r = idx >> 5, c = idx & 31;
      sA[r * BK + c] = A[(size_t)(blockM + r) * K + k0 + c];
    }
#pragma unroll
    for (int i = 0; i < 16; ++i) {
      int idx = tid + i * 256;
      int r = idx >> 5, c = idx & 31;
      sW[r * BK + c] = W[(size_t)(blockN + r) * K + k0 + c];
    }
    if (kt + 1 < kTiles)
      __builtin_prefetch(&A[(size_t)(blockM + (tid >> 1)) * K + k0 + BK], 0, 1);
#endif
    __syncthreads();

    // --- B fragments (4 N-tiles), f32 LDS -> f16 operands ---
    v16h bf[4];
#pragma unroll
    for (int ni = 0; ni < 4; ++ni) {
      const float* bp = &sW[(wn * 64 + ni * 16 + mrow) * BK];
#pragma unroll
      for (int e = 0; e < 16; ++e) bf[ni][e] = (_Float16)bp[kofsA + e + (e >= 8 ? 8 : 0)];
    }
    // --- A fragments (2 M-tiles) x 4 WMMAs each ---
#pragma unroll
    for (int mi = 0; mi < 2; ++mi) {
      v16h af;
      const float* ap = &sA[(wm * 32 + mi * 16 + mrow) * BK];
#pragma unroll
      for (int e = 0; e < 8; ++e) {
        af[e]     = (_Float16)ap[kofsA + e];
        af[e + 8] = (_Float16)ap[kofsA + 16 + e];
      }
#pragma unroll
      for (int ni = 0; ni < 4; ++ni) {
        acc[mi][ni] = __builtin_amdgcn_wmma_f32_16x16x32_f16(
            false, af, false, bf[ni], (short)0, acc[mi][ni], false, false);
      }
    }
    __syncthreads();
  }

  // --- epilogue: bias (+ReLU). C layout: VGPR r -> row rbase+r, col lane&15.
#pragma unroll
  for (int mi = 0; mi < 2; ++mi) {
#pragma unroll
    for (int ni = 0; ni < 4; ++ni) {
      const int col = blockN + wn * 64 + ni * 16 + mrow;
      const float bb = bias[col];
#pragma unroll
      for (int r = 0; r < 8; ++r) {
        const int row = blockM + wm * 32 + mi * 16 + rbase + r;
        float v = acc[mi][ni][r] + bb;
        if (RELU) v = fmaxf(v, 0.0f);
        C[(size_t)row * N + col] = v;
      }
    }
  }
}

// ===========================================================================
// Flash-style masked attention. qkv: [S, 3*D] (q | k | v), out: [S, D].
// Grid: (S/64, H). Block: 128 threads = 4 waves; each wave owns a 16-row
// query tile and streams keys in 32-wide chunks:
//   QK^T : 1 WMMA per 16-key half (hd=32 == one full WMMA K-step)
//   P*V  : LDS bounce C-layout -> A-layout, then 2 WMMAs per chunk
// Online softmax stats in registers, reduced with 16-lane shfls (wave32).
// ===========================================================================
__global__ __launch_bounds__(128) void attention_wmma(
    const float* __restrict__ qkv,  // [S, 3*D_MODEL]
    float* __restrict__ out) {      // [S, D_MODEL]
  const int lane = threadIdx.x & 31;
  const int wave = threadIdx.x >> 5;           // 0..3
  const int head = blockIdx.y;
  const int q0   = blockIdx.x * 64 + wave * 16;
  const int hoff = head * HEAD_DIM;
  const float scale = 0.1767766953f;           // 1/sqrt(32)

  __shared__ _Float16 sP[4][16 * 40];          // per-wave P bounce buffer
  _Float16* pbuf = sP[wave];

  const int kofsA = (lane < 16) ? 0 : 8;
  const int kofsB = (lane < 16) ? 0 : 16;
  const int m = lane & 15;       // A-operand row / B,C-operand column
  const int rbase = (lane < 16) ? 0 : 8;

  // ---- Q fragment (A layout, 16x32), f32 -> f16 on the fly ----
  v16h qf;
  {
    const float* qp = qkv + (size_t)(q0 + m) * (3 * D_MODEL) + hoff;
#pragma unroll
    for (int e = 0; e < 8; ++e) {
      qf[e]     = (_Float16)qp[kofsA + e];
      qf[e + 8] = (_Float16)qp[kofsA + 16 + e];
    }
  }

  float rm[8], rl[8];
#pragma unroll
  for (int r = 0; r < 8; ++r) { rm[r] = -1e30f; rl[r] = 0.0f; }
  v8f o0 = {}, o1 = {};   // output accumulators, head-dims [0,16) / [16,32)

  for (int kc = 0; kc < S_TOT; kc += 32) {
    // prefetch next chunk of K and V rows (global_prefetch_b8)
    if (kc + 32 < S_TOT) {
      __builtin_prefetch(
          &qkv[(size_t)(kc + 32 + lane) * (3 * D_MODEL) + D_MODEL + hoff], 0, 1);
      __builtin_prefetch(
          &qkv[(size_t)(kc + 32 + lane) * (3 * D_MODEL) + 2 * D_MODEL + hoff],
          0, 1);
    }

    // ---- scores for keys [kc, kc+32): two WMMAs against K^T ----
    v8f sc[2];
#pragma unroll
    for (int t = 0; t < 2; ++t) {
      v16h kf;
      const float* kp =
          qkv + (size_t)(kc + t * 16 + m) * (3 * D_MODEL) + D_MODEL + hoff;
#pragma unroll
      for (int e = 0; e < 16; ++e) kf[e] = (_Float16)kp[kofsA + e + (e >= 8 ? 8 : 0)];
      v8f z = {};
      sc[t] = __builtin_amdgcn_wmma_f32_16x16x32_f16(
          false, qf, false, kf, (short)0, z, false, false);
    }

    // ---- scale + additive mask + running softmax ----
#pragma unroll
    for (int r = 0; r < 8; ++r) {
      const int qrow = q0 + rbase + r;
      float best = -1e30f;
#pragma unroll
      for (int t = 0; t < 2; ++t) {
        const int col = kc + t * 16 + m;
        float v = sc[t][r] * scale;
        // text rows never see audio; audio rows are causal over audio
        const bool masked = (col >= X_LEN) && ((qrow < X_LEN) || (col > qrow));
        v += masked ? -100000.0f : 0.0f;
        sc[t][r] = v;
        best = fmaxf(best, v);
      }
#pragma unroll
      for (int d = 1; d < 16; d <<= 1)
        best = fmaxf(best, __shfl_xor(best, d, 32));

      const float mn = fmaxf(rm[r], best);
      const float alpha = __expf(rm[r] - mn);
      float rowsum = 0.0f;
#pragma unroll
      for (int t = 0; t < 2; ++t) {
        const float p = __expf(sc[t][r] - mn);
        sc[t][r] = p;
        rowsum += p;
      }
#pragma unroll
      for (int d = 1; d < 16; d <<= 1) rowsum += __shfl_xor(rowsum, d, 32);
      rl[r] = rl[r] * alpha + rowsum;
      rm[r] = mn;
      o0[r] *= alpha;
      o1[r] *= alpha;
    }

    // ---- bounce P through LDS: C layout -> A layout (same wave, DScnt) ----
#pragma unroll
    for (int r = 0; r < 8; ++r) {
      const int row = rbase + r;
      pbuf[row * 40 + m]      = (_Float16)sc[0][r];
      pbuf[row * 40 + 16 + m] = (_Float16)sc[1][r];
    }
    v16h pf;
    {
      const _Float16* pp = &pbuf[m * 40];
#pragma unroll
      for (int e = 0; e < 8; ++e) {
        pf[e]     = pp[kofsA + e];
        pf[e + 8] = pp[kofsA + 16 + e];
      }
    }

    // ---- P @ V: two WMMAs (head-dim halves) ----
#pragma unroll
    for (int t = 0; t < 2; ++t) {
      v16h vf;
#pragma unroll
      for (int e = 0; e < 16; ++e) {
        const int key = kc + kofsA + e + (e >= 8 ? 8 : 0);
        vf[e] = (_Float16)
            qkv[(size_t)key * (3 * D_MODEL) + 2 * D_MODEL + hoff + t * 16 + m];
      }
      if (t == 0)
        o0 = __builtin_amdgcn_wmma_f32_16x16x32_f16(
            false, pf, false, vf, (short)0, o0, false, false);
      else
        o1 = __builtin_amdgcn_wmma_f32_16x16x32_f16(
            false, pf, false, vf, (short)0, o1, false, false);
    }
  }

  // ---- normalize and write [q0.., hoff..] ----
#pragma unroll
  for (int r = 0; r < 8; ++r) {
    const int row = q0 + rbase + r;
    const float inv = 1.0f / rl[r];
    out[(size_t)row * D_MODEL + hoff + m]      = o0[r] * inv;
    out[(size_t)row * D_MODEL + hoff + 16 + m] = o1[r] * inv;
  }
}

// ===========================================================================
// out[row] = LayerNorm(h[row] + res[row]) * w + b      (one block per row)
// ===========================================================================
__global__ __launch_bounds__(256) void add_layernorm(
    const float* __restrict__ h, const float* __restrict__ res,
    const float* __restrict__ w, const float* __restrict__ b,
    float* __restrict__ out) {
  const int row  = blockIdx.x;
  const int t    = threadIdx.x;
  const int lane = t & 31;
  const int wave = t >> 5;

  const float* hp = h + (size_t)row * D_MODEL;
  const float* rp = res + (size_t)row * D_MODEL;
  const float v0 = hp[t] + rp[t];
  const float v1 = hp[t + 256] + rp[t + 256];

  __shared__ float red[8];

  float s = v0 + v1;
#pragma unroll
  for (int d = 1; d < 32; d <<= 1) s += __shfl_xor(s, d, 32);
  if (lane == 0) red[wave] = s;
  __syncthreads();
  float tot = 0.0f;
#pragma unroll
  for (int i = 0; i < 8; ++i) tot += red[i];
  const float mean = tot * (1.0f / D_MODEL);
  __syncthreads();

  const float d0 = v0 - mean, d1 = v1 - mean;
  s = d0 * d0 + d1 * d1;
#pragma unroll
  for (int d = 1; d < 32; d <<= 1) s += __shfl_xor(s, d, 32);
  if (lane == 0) red[wave] = s;
  __syncthreads();
  tot = 0.0f;
#pragma unroll
  for (int i = 0; i < 8; ++i) tot += red[i];
  const float rstd = rsqrtf(tot * (1.0f / D_MODEL) + LN_EPS);

  out[(size_t)row * D_MODEL + t]       = d0 * rstd * w[t] + b[t];
  out[(size_t)row * D_MODEL + t + 256] = d1 * rstd * w[t + 256] + b[t + 256];
}

// ===========================================================================
// Orchestration
// ===========================================================================
extern "C" void kernel_launch(void* const* d_in, const int* in_sizes, int n_in,
                              void* d_out, int out_size, void* d_ws,
                              size_t ws_size, hipStream_t stream) {
  const float* x      = (const float*)d_in[0];   // [1,S,D]
  const float* qkv_w  = (const float*)d_in[1];   // [L,3D,D]
  const float* qkv_b  = (const float*)d_in[2];   // [L,3D]
  const float* out_w  = (const float*)d_in[3];   // [L,D,D]
  const float* out_b  = (const float*)d_in[4];   // [L,D]
  const float* w1     = (const float*)d_in[5];   // [L,FF,D]
  const float* b1     = (const float*)d_in[6];   // [L,FF]
  const float* w2     = (const float*)d_in[7];   // [L,D,FF]
  const float* b2     = (const float*)d_in[8];   // [L,D]
  const float* nw1    = (const float*)d_in[9];   // [L,D]
  const float* nb1    = (const float*)d_in[10];  // [L,D]
  const float* nw2    = (const float*)d_in[11];  // [L,D]
  const float* nb2    = (const float*)d_in[12];  // [L,D]

  // workspace carve-out (fp32)
  float* ws     = (float*)d_ws;
  float* qkvbuf = ws;                                   // S*3D
  float* attbuf = qkvbuf + (size_t)S_TOT * 3 * D_MODEL; // S*D
  float* obuf   = attbuf + (size_t)S_TOT * D_MODEL;     // S*D
  float* h1buf  = obuf   + (size_t)S_TOT * D_MODEL;     // S*D
  float* m1buf  = h1buf  + (size_t)S_TOT * D_MODEL;     // S*FF
  float* h2buf  = m1buf  + (size_t)S_TOT * FF_DIM;      // S*D

  const dim3 blk256(256), blk128(128);
  const dim3 g_qkv(3 * D_MODEL / BN, S_TOT / BM);   // N=1536
  const dim3 g_out(D_MODEL / BN,     S_TOT / BM);   // N=512
  const dim3 g_ff1(FF_DIM / BN,      S_TOT / BM);   // N=2048
  const dim3 g_att(S_TOT / 64, N_HEAD);

  const float* hin = x;
  for (int l = 0; l < N_LAYER; ++l) {
    const float* qw = qkv_w + (size_t)l * 3 * D_MODEL * D_MODEL;
    const float* qb = qkv_b + (size_t)l * 3 * D_MODEL;
    const float* ow = out_w + (size_t)l * D_MODEL * D_MODEL;
    const float* ob = out_b + (size_t)l * D_MODEL;
    const float* W1 = w1 + (size_t)l * FF_DIM * D_MODEL;
    const float* B1 = b1 + (size_t)l * FF_DIM;
    const float* W2 = w2 + (size_t)l * D_MODEL * FF_DIM;
    const float* B2 = b2 + (size_t)l * D_MODEL;

    // qkv = hin @ qkv_w^T + qkv_b
    gemm_wmma<false><<<g_qkv, blk256, 0, stream>>>(
        hin, qw, qb, qkvbuf, S_TOT, 3 * D_MODEL, D_MODEL);
    // attention
    attention_wmma<<<g_att, blk128, 0, stream>>>(qkvbuf, attbuf);
    // o = att @ out_w^T + out_b
    gemm_wmma<false><<<g_out, blk256, 0, stream>>>(
        attbuf, ow, ob, obuf, S_TOT, D_MODEL, D_MODEL);
    // h1 = LN(hin + o)
    add_layernorm<<<S_TOT, blk256, 0, stream>>>(
        hin, obuf, nw1 + (size_t)l * D_MODEL, nb1 + (size_t)l * D_MODEL, h1buf);
    // m1 = relu(h1 @ w1^T + b1)
    gemm_wmma<true><<<g_ff1, blk256, 0, stream>>>(
        h1buf, W1, B1, m1buf, S_TOT, FF_DIM, D_MODEL);
    // m2 = m1 @ w2^T + b2   (reuse obuf)
    gemm_wmma<false><<<g_out, blk256, 0, stream>>>(
        m1buf, W2, B2, obuf, S_TOT, D_MODEL, FF_DIM);
    // h2 = LN(h1 + m2); final layer writes d_out
    float* hout = (l == N_LAYER - 1) ? (float*)d_out : h2buf;
    add_layernorm<<<S_TOT, blk256, 0, stream>>>(
        h1buf, obuf, nw2 + (size_t)l * D_MODEL, nb2 + (size_t)l * D_MODEL, hout);
    hin = h2buf;
  }
}
